// MSAPairWeightedAveraging_86371792322647
// MI455X (gfx1250) — matrix operand order repaired and checked by reading, not compiled, not run on hardware
//
#include <hip/hip_runtime.h>
#include <hip/hip_bf16.h>

// ---------------- problem constants ----------------
#define S_SEQ   64
#define N_RESI  384
#define C_MDIM  256
#define C_ZDIM  128
#define H_HEADS 8
#define C_HID   32
#define INF_V   1e9f
#define EPS_V   1e-5f

typedef __attribute__((ext_vector_type(16))) _Float16 v16h;
typedef __attribute__((ext_vector_type(8)))  float    v8f;

struct alignas(16) h8 { _Float16 v[8]; };

// ---------------- wave32 reductions ----------------
__device__ __forceinline__ float wave_sum(float v) {
  #pragma unroll
  for (int off = 16; off > 0; off >>= 1) v += __shfl_xor(v, off, 32);
  return v;
}
__device__ __forceinline__ float wave_max(float v) {
  #pragma unroll
  for (int off = 16; off > 0; off >>= 1) v = fmaxf(v, __shfl_xor(v, off, 32));
  return v;
}

// ---------------- WMMA fragment loaders (ISA 7.12.2, wave32) -------------
// A (16x32 f16, row-major source, lda in halves):
//  lanes 0-15: row M=lane, halfs 0-15 -> K=k0+hi*16+t
__device__ __forceinline__ v16h load_frag_a(const _Float16* A, int lda,
                                            int m0, int k0, int lane) {
  int r  = lane & 15;
  int hi = lane >> 4;
  const _Float16* p = A + (size_t)(m0 + r) * lda + k0 + hi * 16;
  v16h a;
  #pragma unroll
  for (int t = 0; t < 8; ++t) a[t] = p[t];
  #pragma unroll
  for (int t = 0; t < 8; ++t) a[8 + t] = p[8 + t];
  return a;
}
// B (32x16 f16) from a K-contiguous (transposed) source Bt[n*ldk + k]:
//  lanes 0-15 hold K=k0..k0+15 (N=lane), lanes 16-31 hold K=k0+16..k0+31 (N=lane-16)
__device__ __forceinline__ v16h load_frag_bt(const _Float16* Bt, int ldk,
                                             int k0, int n0, int lane) {
  int c  = lane & 15;
  int hi = lane >> 4;
  const _Float16* p = Bt + (size_t)(n0 + c) * ldk + k0 + hi * 16;
  v16h b;
  #pragma unroll
  for (int t = 0; t < 16; ++t) b[t] = p[t];
  return b;
}

// ---------------- kernel 1: transpose+convert the three 256x256 weights ----
__global__ void k_prep_weights(const float* __restrict__ Wv,
                               const float* __restrict__ Wg,
                               const float* __restrict__ Wo,
                               _Float16* __restrict__ Wv_t,
                               _Float16* __restrict__ Wg_t,
                               _Float16* __restrict__ Wo_t) {
  int idx = blockIdx.x * blockDim.x + threadIdx.x;
  if (idx >= 3 * C_MDIM * C_MDIM) return;
  int which = idx >> 16;
  int rem   = idx & 65535;
  int k = rem >> 8, n = rem & 255;
  const float* src = (which == 0) ? Wv : (which == 1) ? Wg : Wo;
  _Float16*    dst = (which == 0) ? Wv_t : (which == 1) ? Wg_t : Wo_t;
  dst[n * C_MDIM + k] = (_Float16)src[k * C_MDIM + n];
}

// ---------------- kernel 2: mask bias[j] = INF*(max_s mask[s,j] - 1) -------
__global__ void k_mask_bias(const float* __restrict__ mask, float* __restrict__ bias) {
  int j = threadIdx.x;            // blockDim.x == N_RESI
  float mx = -INF_V;
  for (int s = 0; s < S_SEQ; ++s) mx = fmaxf(mx, mask[s * N_RESI + j]);
  bias[j] = INF_V * (mx - 1.0f);
}

// ---------------- kernel 3: layernorm(m) -> f16, one wave per row ----------
__global__ void k_m_layernorm(const float* __restrict__ m,
                              const float* __restrict__ w,
                              const float* __restrict__ b,
                              _Float16* __restrict__ out) {
  int wave = (blockIdx.x * blockDim.x + threadIdx.x) >> 5;
  int lane = threadIdx.x & 31;
  if (wave >= S_SEQ * N_RESI) return;
  const float* row = m + (size_t)wave * C_MDIM;
  float x[8];
  float s = 0.0f;
  #pragma unroll
  for (int t = 0; t < 8; ++t) { x[t] = row[lane * 8 + t]; s += x[t]; }
  float mu = wave_sum(s) * (1.0f / C_MDIM);
  float var = 0.0f;
  #pragma unroll
  for (int t = 0; t < 8; ++t) { float d = x[t] - mu; var += d * d; }
  var = wave_sum(var) * (1.0f / C_MDIM);
  float rinv = rsqrtf(var + EPS_V);
  h8 o;
  #pragma unroll
  for (int t = 0; t < 8; ++t) {
    int c = lane * 8 + t;
    o.v[t] = (_Float16)((x[t] - mu) * rinv * w[c] + b[c]);
  }
  *(h8*)(out + (size_t)wave * C_MDIM + lane * 8) = o;
}

// ---------------- kernel 4: zb[h,i,j] = (LN(z[i,j,:]) @ W_z)[h], wave/(i,j) -
__global__ void k_pair_bias(const float* __restrict__ z,
                            const float* __restrict__ lw,
                            const float* __restrict__ lb,
                            const float* __restrict__ Wz,
                            float* __restrict__ zb) {
  int wave = (blockIdx.x * blockDim.x + threadIdx.x) >> 5;
  int lane = threadIdx.x & 31;
  if (wave >= N_RESI * N_RESI) return;
  const float* row = z + (size_t)wave * C_ZDIM;
  float x[4];
  float s = 0.0f;
  #pragma unroll
  for (int t = 0; t < 4; ++t) { x[t] = row[lane * 4 + t]; s += x[t]; }
  float mu = wave_sum(s) * (1.0f / C_ZDIM);
  float var = 0.0f;
  #pragma unroll
  for (int t = 0; t < 4; ++t) { float d = x[t] - mu; var += d * d; }
  var = wave_sum(var) * (1.0f / C_ZDIM);
  float rinv = rsqrtf(var + EPS_V);
  float zn[4];
  #pragma unroll
  for (int t = 0; t < 4; ++t) {
    int c = lane * 4 + t;
    zn[t] = (x[t] - mu) * rinv * lw[c] + lb[c];
  }
  #pragma unroll
  for (int h = 0; h < H_HEADS; ++h) {
    float p = 0.0f;
    #pragma unroll
    for (int t = 0; t < 4; ++t) p += zn[t] * Wz[(lane * 4 + t) * H_HEADS + h];
    p = wave_sum(p);
    if (lane == h) zb[(size_t)h * N_RESI * N_RESI + wave] = p;
  }
}

// ---------------- kernel 5: softmax over k -> a f16, one wave per (h,q) ----
__global__ void k_softmax(const float* __restrict__ zb,
                          const float* __restrict__ bias,
                          _Float16* __restrict__ a) {
  int wave = (blockIdx.x * blockDim.x + threadIdx.x) >> 5;
  int lane = threadIdx.x & 31;
  if (wave >= H_HEADS * N_RESI) return;
  const float* row = zb + (size_t)wave * N_RESI;
  float x[12];
  float mx = -INF_V;
  #pragma unroll
  for (int t = 0; t < 12; ++t) {
    int k = lane + 32 * t;
    x[t] = row[k] + bias[k];
    mx = fmaxf(mx, x[t]);
  }
  mx = wave_max(mx);
  float s = 0.0f;
  #pragma unroll
  for (int t = 0; t < 12; ++t) { x[t] = __expf(x[t] - mx); s += x[t]; }
  s = wave_sum(s);
  float r = 1.0f / s;
  #pragma unroll
  for (int t = 0; t < 12; ++t)
    a[(size_t)wave * N_RESI + lane + 32 * t] = (_Float16)(x[t] * r);
}

// ---------------- kernels 6/7/9: WMMA GEMM [24576x256] = A * B[256x256]
// Each wave computes a 16x32 output tile: one A fragment feeds two WMMAs.
// mode 0: store to v_t[(s*H+h)*32+c][k=residue] (K-contiguous for attention B)
// mode 1: sigmoid epilogue, store f16 row-major (gate)
// mode 2: store f32 row-major (final output)
__global__ void k_gemm256(const _Float16* __restrict__ A,
                          const _Float16* __restrict__ Bt,
                          int mode,
                          _Float16* __restrict__ v_t,
                          _Float16* __restrict__ g_h,
                          float* __restrict__ outF) {
  int wave = (blockIdx.x * blockDim.x + threadIdx.x) >> 5;
  int lane = threadIdx.x & 31;
  const int NT2 = C_MDIM / 32;                     // 8 col super-tiles (32 wide)
  const int MT  = (S_SEQ * N_RESI) / 16;           // 1536 row tiles
  if (wave >= MT * NT2) return;
  int mt = wave / NT2, nt = wave % NT2;
  int m0 = mt * 16, n0 = nt * 32;

  v8f acc0 = {}, acc1 = {};
  #pragma unroll
  for (int kk = 0; kk < C_MDIM; kk += 32) {
    v16h af  = load_frag_a(A, C_MDIM, m0, kk, lane);
    v16h bf0 = load_frag_bt(Bt, C_MDIM, kk, n0, lane);
    v16h bf1 = load_frag_bt(Bt, C_MDIM, kk, n0 + 16, lane);
    acc0 = __builtin_amdgcn_wmma_f32_16x16x32_f16(false, af, false, bf0,
                                                  (short)0, acc0, false, false);
    acc1 = __builtin_amdgcn_wmma_f32_16x16x32_f16(false, af, false, bf1,
                                                  (short)0, acc1, false, false);
  }

  int hi = lane >> 4, c16 = lane & 15;
  int mg0 = m0 + 8 * hi;                            // global row base for this lane
  if (mode == 0) {
    // consecutive acc VGPRs are consecutive M rows -> packed f16 stores
    int s = mg0 / N_RESI, nr0 = mg0 % N_RESI;
    #pragma unroll
    for (int half = 0; half < 2; ++half) {
      int j = n0 + half * 16 + c16;
      int h = j >> 5, c = j & 31;
      h8 o;
      #pragma unroll
      for (int t = 0; t < 8; ++t) o.v[t] = (_Float16)(half ? acc1[t] : acc0[t]);
      *(h8*)(v_t + ((size_t)((s * H_HEADS + h) * C_HID + c)) * N_RESI + nr0) = o;
    }
  } else if (mode == 1) {
    int j0 = n0 + c16;
    #pragma unroll
    for (int t = 0; t < 8; ++t) {
      size_t rowb = (size_t)(mg0 + t) * C_MDIM;
      g_h[rowb + j0]      = (_Float16)(1.0f / (1.0f + __expf(-acc0[t])));
      g_h[rowb + j0 + 16] = (_Float16)(1.0f / (1.0f + __expf(-acc1[t])));
    }
  } else {
    int j0 = n0 + c16;
    #pragma unroll
    for (int t = 0; t < 8; ++t) {
      size_t rowb = (size_t)(mg0 + t) * C_MDIM;
      outF[rowb + j0]      = acc0[t];
      outF[rowb + j0 + 16] = acc1[t];
    }
  }
}

// ---------------- kernel 8: attention O = a[h](384x384) @ v[s,h](384x32), gated
// One wave per (s,h,qtile): 16x32 tile covers the full C_HID width.
__global__ void k_attention(const _Float16* __restrict__ a_h,
                            const _Float16* __restrict__ v_t,
                            const _Float16* __restrict__ g_h,
                            _Float16* __restrict__ o_g) {
  int wave = (blockIdx.x * blockDim.x + threadIdx.x) >> 5;
  int lane = threadIdx.x & 31;
  const int QT = N_RESI / 16;                      // 24
  if (wave >= S_SEQ * H_HEADS * QT) return;
  int qt = wave % QT;
  int sh = wave / QT;
  int s = sh >> 3, h = sh & 7;
  int q0 = qt * 16;

  const _Float16* A = a_h + (size_t)h * N_RESI * N_RESI;                 // [q][k]
  const _Float16* B = v_t + (size_t)(s * H_HEADS + h) * C_HID * N_RESI;  // [c][k]

  v8f acc0 = {}, acc1 = {};
  #pragma unroll
  for (int kk = 0; kk < N_RESI; kk += 32) {
    v16h af  = load_frag_a(A, N_RESI, q0, kk, lane);
    v16h bf0 = load_frag_bt(B, N_RESI, kk, 0, lane);
    v16h bf1 = load_frag_bt(B, N_RESI, kk, 16, lane);
    acc0 = __builtin_amdgcn_wmma_f32_16x16x32_f16(false, af, false, bf0,
                                                  (short)0, acc0, false, false);
    acc1 = __builtin_amdgcn_wmma_f32_16x16x32_f16(false, af, false, bf1,
                                                  (short)0, acc1, false, false);
  }

  int hi = lane >> 4, c16 = lane & 15;
  int j0 = h * C_HID + c16;                        // columns j0 and j0+16
  #pragma unroll
  for (int t = 0; t < 8; ++t) {
    int q = q0 + 8 * hi + t;
    size_t rowb = ((size_t)(s * N_RESI + q)) * C_MDIM;
    float og0 = acc0[t] * (float)g_h[rowb + j0];
    float og1 = acc1[t] * (float)g_h[rowb + j0 + 16];
    o_g[rowb + j0]      = (_Float16)og0;
    o_g[rowb + j0 + 16] = (_Float16)og1;
  }
}

// ---------------- host launch ----------------
extern "C" void kernel_launch(void* const* d_in, const int* in_sizes, int n_in,
                              void* d_out, int out_size, void* d_ws, size_t ws_size,
                              hipStream_t stream) {
  (void)in_sizes; (void)n_in; (void)out_size; (void)ws_size;

  const float* m     = (const float*)d_in[0];
  const float* z     = (const float*)d_in[1];
  const float* mask  = (const float*)d_in[2];
  const float* ln_mw = (const float*)d_in[3];
  const float* ln_mb = (const float*)d_in[4];
  const float* ln_zw = (const float*)d_in[5];
  const float* ln_zb = (const float*)d_in[6];
  const float* W_z   = (const float*)d_in[7];
  const float* W_v   = (const float*)d_in[8];
  const float* W_g   = (const float*)d_in[9];
  const float* W_o   = (const float*)d_in[10];
  float* out = (float*)d_out;

  // workspace carve-up (256B aligned slices)
  char* ws = (char*)d_ws;
  size_t off = 0;
  auto take = [&](size_t bytes) -> char* {
    char* p = ws + off;
    off = (off + bytes + 255) & ~(size_t)255;
    return p;
  };
  const size_t ROWS = (size_t)S_SEQ * N_RESI;          // 24576
  float*    bias   = (float*)take(N_RESI * 4);
  _Float16* m_ln_h = (_Float16*)take(ROWS * C_MDIM * 2);
  _Float16* Wv_t   = (_Float16*)take(C_MDIM * C_MDIM * 2);
  _Float16* Wg_t   = (_Float16*)take(C_MDIM * C_MDIM * 2);
  _Float16* Wo_t   = (_Float16*)take(C_MDIM * C_MDIM * 2);
  _Float16* v_t    = (_Float16*)take(ROWS * C_MDIM * 2);
  _Float16* g_h    = (_Float16*)take(ROWS * C_MDIM * 2);
  float*    zb     = (float*)take((size_t)H_HEADS * N_RESI * N_RESI * 4);
  _Float16* a_f16  = (_Float16*)take((size_t)H_HEADS * N_RESI * N_RESI * 2);
  _Float16* o_g    = (_Float16*)take(ROWS * C_MDIM * 2);

  const int gemm_waves  = (int)(ROWS / 16) * (C_MDIM / 32);   // 12288
  const int gemm_blocks = gemm_waves / 8;                     // 1536
  const int attn_waves  = S_SEQ * H_HEADS * (N_RESI / 16);    // 12288
  const int attn_blocks = attn_waves / 8;                     // 1536

  // 1) weights -> transposed f16
  k_prep_weights<<<(3 * C_MDIM * C_MDIM + 255) / 256, 256, 0, stream>>>(
      W_v, W_g, W_o, Wv_t, Wg_t, Wo_t);
  // 2) mask bias
  k_mask_bias<<<1, N_RESI, 0, stream>>>(mask, bias);
  // 3) layernorm(m) -> f16 (one wave per row, 8 waves per block)
  k_m_layernorm<<<(int)(ROWS / 8), 256, 0, stream>>>(m, ln_mw, ln_mb, m_ln_h);
  // 4) pair bias zb[h,i,j] (one wave per (i,j))
  k_pair_bias<<<(N_RESI * N_RESI) / 8, 256, 0, stream>>>(z, ln_zw, ln_zb, W_z, zb);
  // 5) softmax over k -> a f16 (one wave per (h,q))
  k_softmax<<<(H_HEADS * N_RESI) / 8, 256, 0, stream>>>(zb, bias, a_f16);
  // 6) v = m_ln @ W_v   (stored K-contiguous as [s,h,c][k])
  k_gemm256<<<gemm_blocks, 256, 0, stream>>>(m_ln_h, Wv_t, 0, v_t, g_h, out);
  // 7) g = sigmoid(m_ln @ W_g)
  k_gemm256<<<gemm_blocks, 256, 0, stream>>>(m_ln_h, Wg_t, 1, v_t, g_h, out);
  // 8) attention + gating -> o_g f16
  k_attention<<<attn_blocks, 256, 0, stream>>>(a_f16, v_t, g_h, o_g);
  // 9) out = o_g @ W_o (f32)
  k_gemm256<<<gemm_blocks, 256, 0, stream>>>(o_g, Wo_t, 2, v_t, g_h, out);
}
